// stagenet_41540923687310
// MI455X (gfx1250) — matrix operand order repaired and checked by reading, not compiled, not run on hardware
//
#include <hip/hip_runtime.h>
#include <math.h>

// ---- problem constants (B=1) ----
#define Hc   128
#define Wc   160
#define Cch  32
#define Dd   32
#define Vv   4
#define Gg   8
#define HWp  (Hc * Wc)
#define INV_SQRT_C 0.17677669529663687f  // 1/sqrt(32)

typedef __attribute__((ext_vector_type(16))) _Float16 v16h;
typedef __attribute__((ext_vector_type(8)))  float    v8f;

// ---------------------------------------------------------------------------
// Kernel 1: compose projections + affine 4x4 inverse -> per-view rot(9)+trans(3)
// proj layout: (1, V+1, 2, 4, 4): view v: E at v*32, K at v*32+16
// ---------------------------------------------------------------------------
__device__ inline void compose4(const float* E, const float* K, float* P) {
  for (int i = 0; i < 16; ++i) P[i] = E[i];
  for (int i = 0; i < 3; ++i)
    for (int j = 0; j < 4; ++j) {
      float s = 0.f;
      for (int k = 0; k < 3; ++k) s += K[i * 4 + k] * E[k * 4 + j];
      P[i * 4 + j] = s;
    }
}

__global__ void setup_warp(const float* __restrict__ proj, float* __restrict__ wm) {
  if (threadIdx.x != 0 || blockIdx.x != 0) return;
  float refP[16];
  compose4(proj + 0, proj + 16, refP);
  // row3 of refP is [0,0,0,1] by construction -> affine inverse
  const float a0 = refP[0], a1 = refP[1], a2 = refP[2],  t0 = refP[3];
  const float a3 = refP[4], a4 = refP[5], a5 = refP[6],  t1 = refP[7];
  const float a6 = refP[8], a7 = refP[9], a8 = refP[10], t2 = refP[11];
  float i0 = a4 * a8 - a5 * a7;
  float i1 = a2 * a7 - a1 * a8;
  float i2 = a1 * a5 - a2 * a4;
  float i3 = a5 * a6 - a3 * a8;
  float i4 = a0 * a8 - a2 * a6;
  float i5 = a2 * a3 - a0 * a5;
  float i6 = a3 * a7 - a4 * a6;
  float i7 = a1 * a6 - a0 * a7;
  float i8 = a0 * a4 - a1 * a3;
  const float id = 1.f / (a0 * i0 + a1 * i3 + a2 * i6);
  i0 *= id; i1 *= id; i2 *= id; i3 *= id; i4 *= id;
  i5 *= id; i6 *= id; i7 *= id; i8 *= id;
  const float inv[16] = {
      i0, i1, i2, -(i0 * t0 + i1 * t1 + i2 * t2),
      i3, i4, i5, -(i3 * t0 + i4 * t1 + i5 * t2),
      i6, i7, i8, -(i6 * t0 + i7 * t1 + i8 * t2),
      0.f, 0.f, 0.f, 1.f};
  for (int v = 0; v < Vv; ++v) {
    float P[16], PR[16];
    compose4(proj + (v + 1) * 32, proj + (v + 1) * 32 + 16, P);
    for (int i = 0; i < 4; ++i)
      for (int j = 0; j < 4; ++j) {
        float s = 0.f;
        for (int k = 0; k < 4; ++k) s += P[i * 4 + k] * inv[k * 4 + j];
        PR[i * 4 + j] = s;
      }
    float* o = wm + v * 12;
    o[0] = PR[0]; o[1] = PR[1]; o[2] = PR[2];
    o[3] = PR[4]; o[4] = PR[5]; o[5] = PR[6];
    o[6] = PR[8]; o[7] = PR[9]; o[8] = PR[10];
    o[9] = PR[3]; o[10] = PR[7]; o[11] = PR[11];
  }
}

// ---------------------------------------------------------------------------
// Kernel 2: layout transform (n,C,H,W) -> (n,H,W,C) so gathers are float4 loads
// ---------------------------------------------------------------------------
__global__ void transpose_chw_hwc(const float* __restrict__ in,
                                  float* __restrict__ out, int nmaps) {
  const size_t idx = (size_t)blockIdx.x * blockDim.x + threadIdx.x;
  const size_t total = (size_t)nmaps * Cch * HWp;
  if (idx >= total) return;
  const int hw = (int)(idx % HWp);
  const int c = (int)((idx / HWp) % Cch);
  const int m = (int)(idx / ((size_t)HWp * Cch));
  out[((size_t)m * HWp + hw) * Cch + c] = in[idx];
}

// ---------------------------------------------------------------------------
// Kernel 3: main — one wave per pixel; WMMA f16 16x16x32 for grouped correlation
//   A (16x32): M = depth row, K = channel (lanes 0-15: K 0-7,16-23; lanes 16-31:
//              K 8-15,24-31 — ISA 7.12.2 16-bit A layout)
//   B (32x16): block-diagonal ref features; col g = group g channels, col 8 = all
//              channels (gives sum over groups for the per-view softmax)
//   D (16x16): reg r, lane l -> M = r + 8*(l>>4), N = l&15
// ---------------------------------------------------------------------------
template <bool HWC>
__device__ inline float4 ld4(const float* base, int q, int c) {
  if constexpr (HWC) {
    return *(const float4*)(base + (size_t)q * Cch + c);
  } else {
    float4 r;
    r.x = base[(size_t)(c + 0) * HWp + q];
    r.y = base[(size_t)(c + 1) * HWp + q];
    r.z = base[(size_t)(c + 2) * HWp + q];
    r.w = base[(size_t)(c + 3) * HWp + q];
    return r;
  }
}

template <bool HWC>
__global__ __launch_bounds__(256) void stagenet_main(
    const float* __restrict__ refF, const float* __restrict__ srcF,
    const float* __restrict__ depth_hypo, const float* __restrict__ warpmat,
    const float* __restrict__ w_reg, float* __restrict__ out_depth,
    float* __restrict__ conf, float* __restrict__ out_attn) {
  const int lane = threadIdx.x & 31;
  const int p = blockIdx.x * 8 + (threadIdx.x >> 5);  // pixel, grid exact
  if (p >= HWp) return;
  const int half = lane >> 4;  // K-half for A, M-offset half for D
  const int lm = lane & 15;
  const float fx = (float)(p % Wc);
  const float fy = (float)(p / Wc);

  // ---- B: block-diagonal reference features (f16), built once per pixel ----
  v16h bm;
#pragma unroll
  for (int e = 0; e < 16; ++e) {
    const int k = (e < 8) ? (8 * half + e) : (16 + 8 * half + (e - 8));  // B layout matches A: lanes 0-15 K0-7,16-23; 16-31 K8-15,24-31
    float rv;
    if constexpr (HWC) rv = refF[(size_t)p * Cch + k];
    else               rv = refF[(size_t)k * HWp + p];
    const bool on = (lm == (k >> 2)) || (lm == 8);
    bm[e] = on ? (_Float16)rv : (_Float16)0.0f;
  }

  const float zh0 = depth_hypo[(size_t)lm * HWp + p];         // depths 0..15
  const float zh1 = depth_hypo[(size_t)(16 + lm) * HWp + p];  // depths 16..31

  float acc0[8], acc1[8], cs0[8], cs1[8];
#pragma unroll
  for (int r = 0; r < 8; ++r) { acc0[r] = acc1[r] = cs0[r] = cs1[r] = 0.f; }

  const int cA = 8 * half;       // this lane's channel slices for A
  const int cB = 16 + 8 * half;

  for (int v = 0; v < Vv; ++v) {
    const float* wm = warpmat + v * 12;
    const float rx = wm[0] * fx + wm[1] * fy + wm[2];
    const float ry = wm[3] * fx + wm[4] * fy + wm[5];
    const float rz = wm[6] * fx + wm[7] * fy + wm[8];
    const float tx = wm[9], ty = wm[10], tz = wm[11];
    const float* sb = srcF + (size_t)v * HWp * Cch;

    v16h am[2];
#pragma unroll
    for (int dh = 0; dh < 2; ++dh) {
      const float zd = dh ? zh1 : zh0;  // A row M = lm (per half-K lane pair)
      float pz = rz * zd + tz;
      pz = (pz == 0.f) ? 1e-9f : pz;
      float ix = (rx * zd + tx) / pz;
      float iy = (ry * zd + ty) / pz;
      ix = fminf(fmaxf(ix, -1.0e8f), 1.0e8f);
      iy = fminf(fmaxf(iy, -1.0e8f), 1.0e8f);
      const float fx0 = floorf(ix), fy0 = floorf(iy);
      const int x0 = (int)fx0, y0 = (int)fy0;
      const float wx = ix - fx0, wy = iy - fy0;

      float wa[8], wb[8];
#pragma unroll
      for (int e = 0; e < 8; ++e) { wa[e] = 0.f; wb[e] = 0.f; }
#pragma unroll
      for (int corner = 0; corner < 4; ++corner) {
        const int xi = x0 + (corner & 1);
        const int yi = y0 + ((corner >> 1) & 1);
        const float wgt = ((corner & 1) ? wx : (1.f - wx)) *
                          ((corner & 2) ? wy : (1.f - wy));
        const bool valid = (xi >= 0) && (xi < Wc) && (yi >= 0) && (yi < Hc);
        const int q = min(max(yi, 0), Hc - 1) * Wc + min(max(xi, 0), Wc - 1);
        const float w = valid ? wgt : 0.f;
        const float4 g0 = ld4<HWC>(sb, q, cA);
        const float4 g1 = ld4<HWC>(sb, q, cA + 4);
        const float4 g2 = ld4<HWC>(sb, q, cB);
        const float4 g3 = ld4<HWC>(sb, q, cB + 4);
        wa[0] += w * g0.x; wa[1] += w * g0.y; wa[2] += w * g0.z; wa[3] += w * g0.w;
        wa[4] += w * g1.x; wa[5] += w * g1.y; wa[6] += w * g1.z; wa[7] += w * g1.w;
        wb[0] += w * g2.x; wb[1] += w * g2.y; wb[2] += w * g2.z; wb[3] += w * g2.w;
        wb[4] += w * g3.x; wb[5] += w * g3.y; wb[6] += w * g3.z; wb[7] += w * g3.w;
      }
      v16h av;
#pragma unroll
      for (int e = 0; e < 8; ++e) av[e] = (_Float16)wa[e];       // K = e + 8*half
#pragma unroll
      for (int e = 0; e < 8; ++e) av[8 + e] = (_Float16)wb[e];   // K = 16+e+8*half
      am[dh] = av;
    }

    // D[m][g] = sum_{k in group g} warped[m][k]*ref[k] (= 4*cor_feat); col 8 = 4*sum_g
    v8f cz = {};
    v8f d0v = __builtin_amdgcn_wmma_f32_16x16x32_f16(false, am[0], false, bm,
                                                     (short)0, cz, false, false);
    v8f d1v = __builtin_amdgcn_wmma_f32_16x16x32_f16(false, am[1], false, bm,
                                                     (short)0, cz, false, false);

    // per-view softmax over 32 depths of col8/4/ATTN_TEMP = col8 * 0.125
    float mloc = -3.0e38f;
    if (lm == 8) {
#pragma unroll
      for (int r = 0; r < 8; ++r) mloc = fmaxf(mloc, fmaxf(d0v[r], d1v[r]));
    }
    mloc *= 0.125f;
#pragma unroll
    for (int off = 16; off >= 1; off >>= 1)
      mloc = fmaxf(mloc, __shfl_xor(mloc, off, 32));

    float e0[8], e1[8];
    float ssum = 0.f;
#pragma unroll
    for (int r = 0; r < 8; ++r) { e0[r] = 0.f; e1[r] = 0.f; }
    if (lm == 8) {
#pragma unroll
      for (int r = 0; r < 8; ++r) {
        e0[r] = expf(d0v[r] * 0.125f - mloc);
        e1[r] = expf(d1v[r] * 0.125f - mloc);
        ssum += e0[r] + e1[r];
      }
    }
#pragma unroll
    for (int off = 16; off >= 1; off >>= 1) ssum += __shfl_xor(ssum, off, 32);

    const int srcl = (lane & 16) | 8;  // col-8 lane of this half holds my M-rows
    const float sc = INV_SQRT_C / ssum;
#pragma unroll
    for (int r = 0; r < 8; ++r) {
      const float w0 = __shfl(e0[r], srcl, 32) * sc;  // cor_weight, depth r+8*half
      const float w1 = __shfl(e1[r], srcl, 32) * sc;  // depth 16+r+8*half
      cs0[r] += w0;
      cs1[r] += w1;
      acc0[r] += w0 * d0v[r] * 0.25f;  // cor_weight * cor_feat
      acc1[r] += w1 * d1v[r] * 0.25f;
    }
  }

  // ---- normalize, regression logits: sum over group columns within a half ----
  const float wr = (lm < 8) ? w_reg[lm] : 0.f;
  float lg0[8], lg1[8];
#pragma unroll
  for (int r = 0; r < 8; ++r) {
    float t0 = (acc0[r] / (1e-8f + cs0[r])) * wr;
    float t1 = (acc1[r] / (1e-8f + cs1[r])) * wr;
#pragma unroll
    for (int off = 1; off <= 8; off <<= 1) {
      t0 += __shfl_xor(t0, off, 32);
      t1 += __shfl_xor(t1, off, 32);
    }
    lg0[r] = t0;  // logit for depth r+8*half
    lg1[r] = t1;  // logit for depth 16+r+8*half
  }

  // ---- final softmax over 32 depths (halves hold complementary depth sets) ----
  float m2 = -3.0e38f;
#pragma unroll
  for (int r = 0; r < 8; ++r) m2 = fmaxf(m2, fmaxf(lg0[r], lg1[r]));
  m2 = fmaxf(m2, __shfl_xor(m2, 16, 32));
  float s2 = 0.f;
  float ea0[8], ea1[8];
#pragma unroll
  for (int r = 0; r < 8; ++r) {
    ea0[r] = expf(lg0[r] - m2);
    ea1[r] = expf(lg1[r] - m2);
    s2 += ea0[r] + ea1[r];
  }
  s2 += __shfl_xor(s2, 16, 32);
  const float inv2 = 1.f / s2;

  float best = -1.f;
  int bidx = 0;
#pragma unroll
  for (int r = 0; r < 8; ++r) {
    const float a0 = ea0[r] * inv2;
    const float a1 = ea1[r] * inv2;
    const int d0i = r + 8 * half;
    const int d1i = 16 + r + 8 * half;
    if (lm == 0) {  // lanes 0 and 16 cover all 32 depths
      out_attn[(size_t)d0i * HWp + p] = a0;
      out_attn[(size_t)d1i * HWp + p] = a1;
    }
    if (a0 > best || (a0 == best && d0i < bidx)) { best = a0; bidx = d0i; }
    if (a1 > best || (a1 == best && d1i < bidx)) { best = a1; bidx = d1i; }
  }
  {
    const float ob = __shfl_xor(best, 16, 32);
    const int oi = __shfl_xor(bidx, 16, 32);
    if (ob > best || (ob == best && oi < bidx)) { best = ob; bidx = oi; }
  }
  if (lane == 0) {
    out_depth[p] = depth_hypo[(size_t)bidx * HWp + p];
    conf[p] = best;
  }
}

// ---------------------------------------------------------------------------
// Kernel 4: align-corners-style 2x bilinear upsample of conf (stage_idx==2)
// ---------------------------------------------------------------------------
__global__ void upsample_conf(const float* __restrict__ conf,
                              float* __restrict__ out) {
  const int Ho = 2 * Hc, Wo = 2 * Wc;
  const int idx = blockIdx.x * blockDim.x + threadIdx.x;
  if (idx >= Ho * Wo) return;
  const int ox = idx % Wo, oy = idx / Wo;
  const float ys = oy * (float)(Hc - 1) / (float)(Ho - 1);
  const float xs = ox * (float)(Wc - 1) / (float)(Wo - 1);
  int y0 = (int)floorf(ys); y0 = min(max(y0, 0), Hc - 2);
  int x0 = (int)floorf(xs); x0 = min(max(x0, 0), Wc - 2);
  const float wy = ys - y0, wx = xs - x0;
  const float t0 = conf[y0 * Wc + x0] * (1.f - wy) + conf[(y0 + 1) * Wc + x0] * wy;
  const float t1 = conf[y0 * Wc + x0 + 1] * (1.f - wy) + conf[(y0 + 1) * Wc + x0 + 1] * wy;
  out[idx] = t0 * (1.f - wx) + t1 * wx;
}

// ---------------------------------------------------------------------------
extern "C" void kernel_launch(void* const* d_in, const int* in_sizes, int n_in,
                              void* d_out, int out_size, void* d_ws,
                              size_t ws_size, hipStream_t stream) {
  (void)in_sizes; (void)n_in; (void)out_size;
  const float* ref_feat = (const float*)d_in[0];
  const float* src_feats = (const float*)d_in[1];
  const float* depth_hypo = (const float*)d_in[2];
  const float* proj_mats = (const float*)d_in[3];
  const float* w_reg = (const float*)d_in[4];
  // d_in[5] = stage_idx == 2 -> upsample factor 2 (fixed by out_size).

  float* out = (float*)d_out;
  float* out_depth = out;                    // HWp
  float* out_confup = out + HWp;             // 4*HWp
  float* out_attn = out + HWp + 4 * HWp;     // Dd*HWp

  char* ws = (char*)d_ws;
  float* warpmat = (float*)ws;               // 48 floats, pad to 256 B
  float* conf = (float*)(ws + 256);          // HWp floats
  size_t base = (256 + (size_t)HWp * 4 + 255) & ~(size_t)255;
  float* ref_t = (float*)(ws + base);
  float* src_t = ref_t + (size_t)HWp * Cch;
  const size_t need = base + (size_t)(1 + Vv) * HWp * Cch * sizeof(float);

  setup_warp<<<1, 1, 0, stream>>>(proj_mats, warpmat);

  if (ws_size >= need) {
    const int tot1 = Cch * HWp;
    transpose_chw_hwc<<<(tot1 + 255) / 256, 256, 0, stream>>>(ref_feat, ref_t, 1);
    const int tot2 = Vv * Cch * HWp;
    transpose_chw_hwc<<<(tot2 + 255) / 256, 256, 0, stream>>>(src_feats, src_t, Vv);
    stagenet_main<true><<<HWp / 8, 256, 0, stream>>>(
        ref_t, src_t, depth_hypo, warpmat, w_reg, out_depth, conf, out_attn);
  } else {
    stagenet_main<false><<<HWp / 8, 256, 0, stream>>>(
        ref_feat, src_feats, depth_hypo, warpmat, w_reg, out_depth, conf, out_attn);
  }
  upsample_conf<<<(4 * HWp + 255) / 256, 256, 0, stream>>>(conf, out_confup);
}
